// AxialAttentionD_17325898072168
// MI455X (gfx1250) — compile-verified, matrix operand order checked by reading, not run on hardware
//
#include <hip/hip_runtime.h>

// ---------------------------------------------------------------------------
// AxialAttentionD for MI455X (gfx1250, wave32, WMMA f32_16x16x32_f16)
// B=1, C=64, D=48, H=64, W=128, heads=4, head_dim=16
//
// Kernel 1 (fused): per wave = one (head, spatial-pos).
//   - QKV projection via WMMA (A operand shared across Q/K/V, B hoisted)
//   - sinusoidal PE from a per-block f32 LDS table (vector loads, no per-
//     element trig in the hot path)
//   - attn computed TRANSPOSED (attT = K Q^T) so softmax-axis = M dimension:
//     per-lane local reductions + one shfl_xor(16); attT -> LDS becomes
//     r-contiguous b128 stores in row-major As[di][dj]
//   - attn.V via WMMA, staged to workspace as f16 [pos][channel]
// Kernel 2: projection [64x64] x staged -> out, WMMA, weights in LDS,
//   per-wave LDS transpose tile for coalesced 64B output segments.
// ---------------------------------------------------------------------------

typedef __attribute__((ext_vector_type(16))) _Float16 v16h;
typedef __attribute__((ext_vector_type(8)))  _Float16 v8h;
typedef __attribute__((ext_vector_type(8)))  float    v8f;

#define WMMA_F16(a, b, c) \
  __builtin_amdgcn_wmma_f32_16x16x32_f16(false, (a), false, (b), (short)0, (c), false, false)

// Build a 16x32 f16 A-operand (or B^T-operand, same lane mapping with N=row)
// from a row-major f16 tile.  ISA layout (CDNA5 ISA, 16-bit A 16x32):
//   lane L: m = L&15, kb = (L>>4)*8
//   VGPR p (half q): K = p<4 ? kb+2p+q : 16+kb+2(p-4)+q
// K-padding: validity is lane-uniform in all uses, so the bound uses the
// lane-maximum kb (=8) -> compile-time constant after unrolling.
__device__ __forceinline__ v16h load_tile_f16(const _Float16* __restrict__ base,
                                              int row_stride, int mbase,
                                              int ko, int kvalid, int lane) {
  const int m  = lane & 15;
  const int kb = (lane >> 4) << 3;
  const _Float16* row = base + (mbase + m) * row_stride;
  v16h a;
#pragma unroll
  for (int p = 0; p < 8; ++p) {
#pragma unroll
    for (int q = 0; q < 2; ++q) {
      const int k    = (p < 4) ? (kb + 2 * p + q) : (16 + kb + 2 * (p - 4) + q);
      const int kmax = (p < 4) ? (8 + 2 * p + q)  : (24 + 2 * (p - 4) + q);
      a[2 * p + q] = ((ko + kmax) < kvalid) ? row[ko + k] : (_Float16)0.0f;
    }
  }
  return a;
}

__global__ __launch_bounds__(256)
void axial_qkv_attn_kernel(const float* __restrict__ x,
                           const float* __restrict__ qkv_w,
                           _Float16* __restrict__ staged) {
  constexpr int Dd = 48, HW = 64 * 128, DHW = Dd * HW;
  constexpr float SCALE = 0.25f;                 // head_dim^-0.5
  constexpr float LOG1E4_OVER_7 = 1.3157629103f; // ln(10000)/7
  constexpr float HALF_PI = 1.5707963268f;

  // LDS carve-up (dynamic), in halves (Pe is float, appended at the end):
  extern __shared__ _Float16 sm[];
  _Float16* Wl = sm;                    // 192*72 qkv weights (row-major, pad 72)
  _Float16* Xs = Wl + 192 * 72;         // 2*48*72 x tile  [s2][d][c]
  _Float16* Qs = Xs + 2 * 48 * 72;      // 8 slots * 48*16 [d][c]  (PE+scale)
  _Float16* Ks = Qs + 8 * 48 * 16;      // 8 slots * 48*16 [d][c]  (PE)
  _Float16* Vt = Ks + 8 * 48 * 16;      // 8 slots * 16*48 [c][d]  (transposed)
  _Float16* As = Vt + 8 * 16 * 48;      // 8 slots * 48*64 softmaxed attn [di][dj]
  float*    Pe = (float*)(As + 8 * 48 * 64);  // 16*48 f32 PE table [n][d]

  const int tid   = threadIdx.x;
  const int lane  = tid & 31;
  const int wid   = tid >> 5;           // 8 waves
  const int sbase = blockIdx.x * 2;     // 2 spatial positions per block

  // ---- cooperative stage: weights, x tile, PE table ----
  for (int i = tid; i < 192 * 64; i += 256) {
    const int r = i >> 6, c = i & 63;
    Wl[r * 72 + c] = (_Float16)qkv_w[i];
  }
  for (int i = tid; i < 2 * 48 * 64; i += 256) {
    const int s2 = i & 1;
    const int d  = (i >> 1) % 48;
    const int c  = (i >> 1) / 48;
    Xs[(s2 * 48 + d) * 72 + c] = (_Float16)x[c * DHW + d * HW + sbase + s2];
  }
  for (int i = tid; i < 16 * 48; i += 256) {     // pe(n,d)=sin(d*invf + phase)
    const int pn = i / 48, pd = i % 48;
    const float invf  = __expf(-(float)(pn >> 1) * LOG1E4_OVER_7);
    const float phase = (pn & 1) ? HALF_PI : 0.0f;
    Pe[pn * 48 + pd] = __sinf((float)pd * invf + phase);
  }
  __syncthreads();

  // wave -> (head, local spatial)
  const int hq   = wid >> 1;
  const int sl   = wid & 1;
  const int slot = wid;
  const int n    = lane & 15;           // N column owned by this lane (C/D layout)
  const int hi   = lane >> 4;           // M-half

  const _Float16* Xrow = Xs + sl * 48 * 72;

  // ---- QKV projection: D[d][cdim] = sum_c x[c,d,s] * w[row+cdim][c] ----
  const _Float16* WQ = Wl + (0 * 64 + hq * 16) * 72;
  const _Float16* WK = Wl + (1 * 64 + hq * 16) * 72;
  const _Float16* WV = Wl + (2 * 64 + hq * 16) * 72;
  const v16h bQ0 = load_tile_f16(WQ, 72, 0, 0,  64, lane);
  const v16h bQ1 = load_tile_f16(WQ, 72, 0, 32, 64, lane);
  const v16h bK0 = load_tile_f16(WK, 72, 0, 0,  64, lane);
  const v16h bK1 = load_tile_f16(WK, 72, 0, 32, 64, lane);
  const v16h bV0 = load_tile_f16(WV, 72, 0, 0,  64, lane);
  const v16h bV1 = load_tile_f16(WV, 72, 0, 32, 64, lane);

#pragma unroll
  for (int mt = 0; mt < 3; ++mt) {                // d-tiles (48 = 3*16)
    const v16h a0 = load_tile_f16(Xrow, 72, mt * 16, 0,  64, lane);
    const v16h a1 = load_tile_f16(Xrow, 72, mt * 16, 32, 64, lane);
    v8f aq = {}, ak = {}, av = {};
    aq = WMMA_F16(a0, bQ0, aq);  aq = WMMA_F16(a1, bQ1, aq);
    ak = WMMA_F16(a0, bK0, ak);  ak = WMMA_F16(a1, bK1, ak);
    av = WMMA_F16(a0, bV0, av);  av = WMMA_F16(a1, bV1, av);

    // PE for this lane: fixed n, 8 consecutive d -> vector LDS load
    const float* pe_row = Pe + n * 48 + mt * 16 + hi * 8;
    v8h pv;
#pragma unroll
    for (int r = 0; r < 8; ++r) {
      const int d  = mt * 16 + r + hi * 8;
      const float pe = pe_row[r];
      Qs[(slot * 48 + d) * 16 + n] = (_Float16)((aq[r] + pe) * SCALE);
      Ks[(slot * 48 + d) * 16 + n] = (_Float16)(ak[r] + pe);
      pv[r] = (_Float16)av[r];
    }
    // V transposed: d contiguous per lane -> one b128 store
    *(v8h*)(Vt + (slot * 16 + n) * 48 + mt * 16 + hi * 8) = pv;
  }
  // Qs/Ks/Vt produced and consumed by the same wave: no block barrier.

  // ---- attT = K Q^T (transposed attention; K-dim = 16, zero-padded) ----
  // Tile(jt,it): A = K rows [jt*16..], B^T = Q rows [it*16..]
  // C/D element: m = dj = jt*16 + r + hi*8 (softmax axis!), n = di = it*16 + n
  v8f att[3][3];                        // [jt][it]
  {
    const v16h bq0 = load_tile_f16(Qs + slot * 48 * 16, 16, 0,  0, 16, lane);
    const v16h bq1 = load_tile_f16(Qs + slot * 48 * 16, 16, 16, 0, 16, lane);
    const v16h bq2 = load_tile_f16(Qs + slot * 48 * 16, 16, 32, 0, 16, lane);
#pragma unroll
    for (int jt = 0; jt < 3; ++jt) {
      const v16h akk = load_tile_f16(Ks + slot * 48 * 16, 16, jt * 16, 0, 16, lane);
      v8f z0 = {}, z1 = {}, z2 = {};
      att[jt][0] = WMMA_F16(akk, bq0, z0);
      att[jt][1] = WMMA_F16(akk, bq1, z1);
      att[jt][2] = WMMA_F16(akk, bq2, z2);
    }
  }

  // ---- softmax over dj: local reduce over 24 regs + one shfl_xor(16) ----
#pragma unroll
  for (int it = 0; it < 3; ++it) {      // each it = a different di per lane
    float mx = att[0][it][0];
#pragma unroll
    for (int jt = 0; jt < 3; ++jt)
#pragma unroll
      for (int r = 0; r < 8; ++r) mx = fmaxf(mx, att[jt][it][r]);
    mx = fmaxf(mx, __shfl_xor(mx, 16, 32));

    float sum = 0.0f;
#pragma unroll
    for (int jt = 0; jt < 3; ++jt)
#pragma unroll
      for (int r = 0; r < 8; ++r) {
        const float e = __expf(att[jt][it][r] - mx);
        att[jt][it][r] = e;
        sum += e;
      }
    sum += __shfl_xor(sum, 16, 32);
    const float rcp = __builtin_amdgcn_rcpf(sum);
#pragma unroll
    for (int jt = 0; jt < 3; ++jt)
#pragma unroll
      for (int r = 0; r < 8; ++r) att[jt][it][r] *= rcp;
  }

  // attT -> row-major As[di][dj]: dj contiguous per lane -> b128 stores
#pragma unroll
  for (int it = 0; it < 3; ++it) {
#pragma unroll
    for (int jt = 0; jt < 3; ++jt) {
      v8h pa;
#pragma unroll
      for (int r = 0; r < 8; ++r) pa[r] = (_Float16)att[jt][it][r];
      *(v8h*)(As + (slot * 48 + it * 16 + n) * 64 + jt * 16 + hi * 8) = pa;
    }
  }

  // ---- out = attn V  (K-dim = 48, compile-time zero-padded to 64) ----
  const int sg = sbase + sl;
  const v16h bv0 = load_tile_f16(Vt + slot * 16 * 48, 48, 0, 0,  48, lane);
  const v16h bv1 = load_tile_f16(Vt + slot * 16 * 48, 48, 0, 32, 48, lane);
#pragma unroll
  for (int mt = 0; mt < 3; ++mt) {
    v8f acc = {};
    acc = WMMA_F16(load_tile_f16(As + slot * 48 * 64, 64, mt * 16, 0,  48, lane), bv0, acc);
    acc = WMMA_F16(load_tile_f16(As + slot * 48 * 64, 64, mt * 16, 32, 48, lane), bv1, acc);
#pragma unroll
    for (int r = 0; r < 8; ++r) {
      const int d = mt * 16 + r + hi * 8;
      staged[(size_t)(d * HW + sg) * 64 + hq * 16 + n] = (_Float16)acc[r];
    }
  }
}

__global__ __launch_bounds__(256)
void axial_proj_kernel(const _Float16* __restrict__ staged,
                       const float* __restrict__ proj_w,
                       float* __restrict__ out) {
  constexpr int DHW = 48 * 8192;
  __shared__ _Float16 Wp[64 * 72];
  __shared__ float    Ot[8][64 * 17];   // per-wave transpose tile, pad 17

  const int tid = threadIdx.x, lane = tid & 31, wid = tid >> 5;
  for (int i = tid; i < 64 * 64; i += 256) {
    const int r = i >> 6, c = i & 63;
    Wp[r * 72 + c] = (_Float16)proj_w[i];
  }
  __syncthreads();

  const int pbase = blockIdx.x * 128 + wid * 16;   // 16 positions per wave
  const int n = lane & 15, hi = lane >> 4;

  const v16h a0 = load_tile_f16(staged + (size_t)pbase * 64, 64, 0, 0,  64, lane);
  const v16h a1 = load_tile_f16(staged + (size_t)pbase * 64, 64, 0, 32, 64, lane);

#pragma unroll
  for (int nt = 0; nt < 4; ++nt) {                 // 64 out channels = 4 tiles
    const v16h b0 = load_tile_f16(Wp + nt * 16 * 72, 72, 0, 0,  64, lane);
    const v16h b1 = load_tile_f16(Wp + nt * 16 * 72, 72, 0, 32, 64, lane);
    v8f acc = {};
    acc = WMMA_F16(a0, b0, acc);
    acc = WMMA_F16(a1, b1, acc);
#pragma unroll
    for (int r = 0; r < 8; ++r)                    // C/D layout -> LDS transpose
      Ot[wid][(nt * 16 + n) * 17 + r + hi * 8] = acc[r];
  }

  // same-wave consume: coalesced 64B row segments to global
#pragma unroll 4
  for (int i = lane; i < 64 * 16; i += 32) {
    const int o = i >> 4, j = i & 15;
    out[(size_t)o * DHW + pbase + j] = Ot[wid][o * 17 + j];
  }
}

extern "C" void kernel_launch(void* const* d_in, const int* in_sizes, int n_in,
                              void* d_out, int out_size, void* d_ws, size_t ws_size,
                              hipStream_t stream) {
  const float* x      = (const float*)d_in[0];   // [1,64,48,64,128]
  const float* qkv_w  = (const float*)d_in[1];   // [192,64]
  const float* proj_w = (const float*)d_in[2];   // [64,64]
  float*       out    = (float*)d_out;           // [1,64,48,64,128]
  _Float16*    staged = (_Float16*)d_ws;         // [48*8192][64] f16, ~50.3 MB

  const size_t smem = (size_t)(192 * 72 + 2 * 48 * 72 +
                               8 * 48 * 16 * 2 + 8 * 16 * 48 + 8 * 48 * 64) *
                          sizeof(_Float16) +
                      (size_t)(16 * 48) * sizeof(float);   // ~127.5 KB

  // 4096 blocks: 2 spatial positions x 4 heads per block (8 waves)
  axial_qkv_attn_kernel<<<4096, 256, smem, stream>>>(x, qkv_w, staged);
  // 3072 blocks: 128 positions per block
  axial_proj_kernel<<<3072, 256, 0, stream>>>(staged, proj_w, out);
}
